// Softmaxchain_74406013435918
// MI455X (gfx1250) — compile-verified, matrix operand order checked by reading, not run on hardware
//
#include <hip/hip_runtime.h>

#define ATTN_H 16
#define ATTN_S 2048
#define ATTN_D 64

typedef __attribute__((ext_vector_type(16))) __bf16         v16bf;
typedef __attribute__((ext_vector_type(8)))  float          v8f;
typedef __attribute__((ext_vector_type(16))) unsigned short v16u;
typedef __attribute__((ext_vector_type(8)))  unsigned short v8u;
typedef __attribute__((ext_vector_type(2)))  float          f32x2;
typedef __attribute__((ext_vector_type(2)))  __bf16         bf16x2;

// Two fp32 -> packed bf16 (RNE); gfx1250 selects v_cvt_pk_bf16_f32.
static __device__ __forceinline__ unsigned int pk_bf16(float a, float b) {
  f32x2 f; f[0] = a; f[1] = b;
  bf16x2 h = __builtin_convertvector(f, bf16x2);
  union { bf16x2 h; unsigned int u; } c; c.h = h;
  return c.u;
}
static __device__ __forceinline__ unsigned short f2bf(float a) {
  union { __bf16 h; unsigned short u; } c; c.h = (__bf16)a;
  return c.u;
}
static __device__ __forceinline__ v16bf as_bf16(v16u x) {
  union { v16u u; v16bf b; } c; c.u = x; return c.b;
}
static __device__ __forceinline__ v16u cat8(v8u lo, v8u hi) {
  return __builtin_shufflevector(lo, hi, 0,1,2,3,4,5,6,7,8,9,10,11,12,13,14,15);
}
// One 16-bit A/B fragment = two aligned ds_load_b128 (elems 0..7 = K base..base+7,
// elems 8..15 = K base+16..base+23 per the CDNA5 16-bit VGPR layout).
static __device__ __forceinline__ v16bf ld_frag(const unsigned short* p) {
  return as_bf16(cat8(*(const v8u*)p, *(const v8u*)(p + 16)));
}

__global__ __launch_bounds__(256, 1)
void Softmaxchain_74406013435918_kernel(const float* __restrict__ q,
                                        const float* __restrict__ k,
                                        const float* __restrict__ v,
                                        const float* __restrict__ mask,
                                        float* __restrict__ out)
{
  // LDS staging (bf16 bits in u16). Row strides are multiples of 8 u16 (16B)
  // so every v8u fragment read is one aligned ds_load_b128.
  __shared__ __align__(16) unsigned short lKt[64][72];    // K blk, [t][d] d-contig
  __shared__ __align__(16) unsigned short lVt[64][72];    // V blk, [d][t] t-contig
  __shared__ __align__(16) unsigned short lP [8][16][72]; // per-wave P, [m][t]

  const int tid  = threadIdx.x;
  const int wave = tid >> 5;
  const int lane = tid & 31;
  const int bh   = blockIdx.x >> 4;          // b*H + h
  const int qblk = blockIdx.x & 15;          // 128-row q block
  const int bidx = bh >> 4;                  // batch index (H == 16)
  const int qbase = qblk * 128 + wave * 16;

  const float* __restrict__ qh = q + (size_t)bh * ATTN_S * ATTN_D;
  const float* __restrict__ kh = k + (size_t)bh * ATTN_D * ATTN_S;  // [d][t]
  const float* __restrict__ vh = v + (size_t)bh * ATTN_S * ATTN_D;  // [t][d]
  const float* __restrict__ mb = mask + (size_t)bidx * ATTN_S;
  float* __restrict__ oh = out + (size_t)bh * ATTN_S * ATTN_D;

  const int mrow  = lane & 15;             // A-row / B-col owned by this lane
  const int kb    = (lane < 16) ? 0 : 8;   // K-chunk base of 16-bit A/B layout
  const int rbase = (lane < 16) ? 0 : 8;   // C/D rows owned by this lane-half
  const int tcol  = lane & 15;

  // ---- one-time: this wave's 16x64 Q tile as two bf16 A fragments
  v16bf aq[2];
  {
    const float* qrow = qh + (size_t)(qbase + mrow) * ATTN_D;
    #pragma unroll
    for (int f = 0; f < 2; ++f) {
      v16u u;
      #pragma unroll
      for (int p = 0; p < 8; ++p) {
        int d = f * 32 + ((p < 4) ? (kb + 2 * p) : (16 + kb + 2 * (p - 4)));
        u[2 * p + 0] = f2bf(qrow[d + 0]);
        u[2 * p + 1] = f2bf(qrow[d + 1]);
      }
      aq[f] = as_bf16(u);
    }
  }

  v8f acc[4];
  float m_i[8], l_i[8];                    // l_i is a PER-LANE partial sum
  #pragma unroll
  for (int j = 0; j < 4; ++j)
    #pragma unroll
    for (int r = 0; r < 8; ++r) acc[j][r] = 0.0f;
  #pragma unroll
  for (int r = 0; r < 8; ++r) { m_i[r] = -__builtin_inff(); l_i[r] = 0.0f; }

  const float L2E = 1.44269504088896f;

  for (int t0 = 0; t0 < ATTN_S; t0 += 64) {
    // ---- cooperative stage, fp32->bf16, packed ds_store_b32, coalesced reads
    #pragma unroll
    for (int i = 0; i < 8; ++i) {          // K: 64x64, pack pairs along d
      int idx = tid + i * 256;             // 0..2047
      int d = (idx >> 6) * 2, t = idx & 63;
      unsigned int p = pk_bf16(kh[(size_t)d * ATTN_S + t0 + t],
                               kh[(size_t)(d + 1) * ATTN_S + t0 + t]);
      *(unsigned int*)&lKt[t][d] = p;
    }
    #pragma unroll
    for (int i = 0; i < 8; ++i) {          // V: 64x64, pack pairs along t
      int idx = tid + i * 256;
      int d = idx & 63, t = (idx >> 6) * 2;
      unsigned int p = pk_bf16(vh[(size_t)(t0 + t) * ATTN_D + d],
                               vh[(size_t)(t0 + t + 1) * ATTN_D + d]);
      *(unsigned int*)&lVt[d][t] = p;
    }
    __syncthreads();

    // ---- prefetch next K/V block while we compute this one
    if (t0 + 64 < ATTN_S) {
      const int t0n = t0 + 64;
      __builtin_prefetch(kh + (size_t)(tid >> 2) * ATTN_S + t0n + (tid & 3) * 16, 0, 3);
      __builtin_prefetch(vh + (size_t)(t0n + (tid >> 2)) * ATTN_D + (tid & 3) * 16, 0, 3);
    }

    // ---- S[16x64] = Q[16x64] @ K[64x64]
    // Preload ALL 8 K fragments (16 ds_load_b128, distinct regs), then run the
    // 8 WMMAs back-to-back (C-accumulation chains carry no RAW hazard NOPs).
    v16bf bk[8];
    #pragma unroll
    for (int u = 0; u < 8; ++u)
      bk[u] = ld_frag(&lKt[(u & 3) * 16 + tcol][(u >> 2) * 32 + kb]);

    v8f s[4];
    #pragma unroll
    for (int j = 0; j < 4; ++j)
      #pragma unroll
      for (int r = 0; r < 8; ++r) s[j][r] = 0.0f;
    #pragma unroll
    for (int u = 0; u < 8; ++u)
      s[u & 3] = __builtin_amdgcn_wmma_f32_16x16x32_bf16(false, aq[u >> 2], false, bk[u],
                                                         (short)0, s[u & 3], false, false);

    // ---- hoist the 8 V fragments now: their LDS latency hides under softmax
    v16bf bv[8];
    #pragma unroll
    for (int u = 0; u < 8; ++u)
      bv[u] = ld_frag(&lVt[(u & 3) * 16 + tcol][(u >> 2) * 32 + kb]);

    // ---- scale 1/8 + additive mask
    float mk[4];
    #pragma unroll
    for (int j = 0; j < 4; ++j) mk[j] = mb[t0 + j * 16 + tcol];
    #pragma unroll
    for (int j = 0; j < 4; ++j)
      #pragma unroll
      for (int r = 0; r < 8; ++r) s[j][r] = s[j][r] * 0.125f + mk[j];

    // ---- online softmax: only MAX needs a butterfly; row-sum stays per-lane
    #pragma unroll
    for (int r = 0; r < 8; ++r) {
      float mx = fmaxf(fmaxf(s[0][r], s[1][r]), fmaxf(s[2][r], s[3][r]));
      #pragma unroll
      for (int off = 8; off >= 1; off >>= 1) mx = fmaxf(mx, __shfl_xor(mx, off, 32));
      float mnew  = fmaxf(m_i[r], mx);
      float alpha = exp2f((m_i[r] - mnew) * L2E);
      m_i[r] = mnew;
      float p0 = exp2f((s[0][r] - mnew) * L2E);
      float p1 = exp2f((s[1][r] - mnew) * L2E);
      float p2 = exp2f((s[2][r] - mnew) * L2E);
      float p3 = exp2f((s[3][r] - mnew) * L2E);
      s[0][r] = p0; s[1][r] = p1; s[2][r] = p2; s[3][r] = p3;
      l_i[r] = l_i[r] * alpha + ((p0 + p1) + (p2 + p3));   // lane-local partial
      acc[0][r] *= alpha; acc[1][r] *= alpha;
      acc[2][r] *= alpha; acc[3][r] *= alpha;
    }

    // ---- C-layout P -> A-layout P via per-wave LDS bounce.
    // Same-wave DS ops are processed in order, so the RAW through lP is safe;
    // the compiler inserts the dscnt wait for the loaded VGPRs before use.
    #pragma unroll
    for (int r = 0; r < 8; ++r)
      #pragma unroll
      for (int j = 0; j < 4; ++j)
        lP[wave][rbase + r][j * 16 + tcol] = f2bf(s[j][r]);
    v16bf ap[2];
    #pragma unroll
    for (int ks = 0; ks < 2; ++ks)
      ap[ks] = ld_frag(&lP[wave][mrow][ks * 32 + kb]);

    // ---- context += P[16x64] @ V[64x64]: 8 back-to-back WMMAs (bv preloaded)
    #pragma unroll
    for (int u = 0; u < 8; ++u)
      acc[u & 3] = __builtin_amdgcn_wmma_f32_16x16x32_bf16(false, ap[u >> 2], false, bv[u],
                                                           (short)0, acc[u & 3], false, false);
    __syncthreads();
  }

  // ---- epilogue: finish the deferred row-sum reduction, normalize, store
  #pragma unroll
  for (int r = 0; r < 8; ++r) {
    float ls = l_i[r];
    #pragma unroll
    for (int off = 8; off >= 1; off >>= 1) ls += __shfl_xor(ls, off, 32);
    float inv = 1.0f / ls;
    float* orow = oh + (size_t)(qbase + rbase + r) * ATTN_D;
    #pragma unroll
    for (int j = 0; j < 4; ++j) orow[j * 16 + tcol] = acc[j][r] * inv;
  }
}

extern "C" void kernel_launch(void* const* d_in, const int* in_sizes, int n_in,
                              void* d_out, int out_size, void* d_ws, size_t ws_size,
                              hipStream_t stream) {
  (void)in_sizes; (void)n_in; (void)d_ws; (void)ws_size; (void)out_size;
  const float* q    = (const float*)d_in[0];
  const float* k    = (const float*)d_in[1];
  const float* v    = (const float*)d_in[2];
  const float* mask = (const float*)d_in[3];
  float* out        = (float*)d_out;

  // 64 heads x (2048/128) q-blocks = 1024 workgroups of 256 threads (8 waves)
  dim3 grid(4 * ATTN_H * (ATTN_S / 128));
  dim3 block(256);
  Softmaxchain_74406013435918_kernel<<<grid, block, 0, stream>>>(q, k, v, mask, out);
}
